// SMC_Transf_Cell_9637906612541
// MI455X (gfx1250) — matrix operand order repaired and checked by reading, not compiled, not run on hardware
//
#include <hip/hip_runtime.h>
#include <hip/hip_bf16.h>

#define DEV __device__ __forceinline__

typedef __attribute__((ext_vector_type(16))) __bf16 v16bf;
typedef __attribute__((ext_vector_type(8)))  float  v8f;

// ---- reference constants ----
constexpr int B_ = 16, P_ = 16, S_ = 512, D_ = 256, DFF_ = 1024, VOCAB_ = 5000, T_ = 256;
constexpr int BP = B_ * P_;                 // 256 token rows
constexpr float EPS_  = 1e-6f;
constexpr float SCALE = 0.0625f;            // 1/sqrt(256)

// ---- output layout (tuple concat: r, attn, noise_q, noise_z, K, V, R) ----
constexpr size_t OFF_R    = 0;
constexpr size_t OFF_ATTN = OFF_R    + (size_t)BP * D_;            // 65536
constexpr size_t OFF_NQ   = OFF_ATTN + (size_t)BP * (T_ + 1);      // 131328
constexpr size_t OFF_NZ   = OFF_NQ   + (size_t)BP * D_;            // 196864
constexpr size_t OFF_K    = OFF_NZ   + (size_t)BP * D_;            // 262400
constexpr size_t OFF_V    = OFF_K    + (size_t)BP * S_ * D_;       // 33816832
constexpr size_t OFF_RR   = OFF_V    + (size_t)BP * S_ * D_;       // 67371264

// ---- workspace layout (floats); total ~8.0 MB ----
constexpr size_t WS_Q    = 0;
constexpr size_t WS_KS   = WS_Q    + (size_t)BP * D_;
constexpr size_t WS_VS   = WS_KS   + (size_t)BP * D_;
constexpr size_t WS_H    = WS_VS   + (size_t)BP * D_;
constexpr size_t WS_Z    = WS_H    + (size_t)BP * D_;
constexpr size_t WS_O1   = WS_Z    + (size_t)BP * D_;
constexpr size_t WS_F1   = WS_O1   + (size_t)BP * D_;
constexpr size_t WS_F2   = WS_F1   + (size_t)BP * DFF_;
constexpr size_t WS_PRED = WS_F2   + (size_t)BP * D_;
constexpr size_t WS_W    = WS_PRED + (size_t)BP * VOCAB_;
constexpr size_t WS_IT   = WS_W    + 256;   // int region

// ---------------- deterministic device RNG (hash -> Box-Muller) ----------------
DEV uint32_t hmix(uint32_t x) {
  x ^= x >> 16; x *= 0x7feb352du;
  x ^= x >> 15; x *= 0x846ca68bu;
  x ^= x >> 16; return x;
}
DEV float u01(uint32_t x) { return ((float)(x >> 8) + 0.5f) * (1.0f / 16777216.0f); }
DEV float gaussian(uint32_t idx, uint32_t salt) {
  uint32_t h1 = hmix(idx * 0x9E3779B9u + salt * 0x85EBCA6Bu + 1u);
  uint32_t h2 = hmix(h1 + 0xC2B2AE35u);
  return sqrtf(-2.0f * __logf(u01(h1))) * __cosf(6.28318530718f * u01(h2));
}

// ---------------- WMMA bf16 GEMM: C[MxN] = A[MxK] @ W[KxN] (+bias, opt. relu) ----------------
// 4 waves/block; each wave owns a 16x32 output strip (two C tiles sharing one
// A fragment -> 2 v_wmma per K-chunk). K and N are compile-time so every load
// in the chunk loop is a constant-immediate offset off a walking base pointer.
// A is loaded as four explicit float4 (global_load_b128); B as 16 coalesced
// b32 per tile.
// Lane layouts used here (contiguous k order, not the ISA 7.12.2 table order):
//   A(16x32 bf16): lane l15 holds row; element e -> K = 16*h + e.
//   B(32x16 bf16): lane l15 holds column;  element e -> K = 16*h + e.
//   C/D(16x16 f32): row = i + 8*h, col = l15.
// OOB columns only need ADDRESS clamping: a garbage B column only feeds the
// same (unstored) D column.
template<int K, int N>
__global__ void gemm_wmma_bf16(const float* __restrict__ A, const float* __restrict__ W,
                               const float* __restrict__ bias, float* __restrict__ C,
                               int relu)
{
  const int lane = threadIdx.x & 31;
  const int wave = threadIdx.x >> 5;
  const int h    = lane >> 4;           // half-wave select
  const int l15  = lane & 15;
  const int m0   = blockIdx.y * 16;
  const int n0   = blockIdx.x * 128 + wave * 32;
  if (n0 >= N) return;                  // wave-uniform exit (EXEC stays full for WMMA)

  const int c0 = n0 + l15;              // columns of the two tiles
  const int c1 = c0 + 16;
  const int c0cl = (c0 < N) ? c0 : (N - 1);
  const int c1cl = (c1 < N) ? c1 : (N - 1);

  v8f acc0 = {0.f,0.f,0.f,0.f,0.f,0.f,0.f,0.f};
  v8f acc1 = {0.f,0.f,0.f,0.f,0.f,0.f,0.f,0.f};

  const float* pA  = A + (size_t)(m0 + l15) * K + h * 16;  // contiguous span +0..15
  const float* pB0 = W + (size_t)(h * 16) * N + c0cl;      // offsets e*N (constant)
  const float* pB1 = W + (size_t)(h * 16) * N + c1cl;

#pragma unroll 2
  for (int kk = 0; kk < K; kk += 32) {
    // A fragment: 16 contiguous floats per lane (K = 16*h + e) -> 4x b128
    const float4 a0 = *(const float4*)(pA + 0);
    const float4 a1 = *(const float4*)(pA + 4);
    const float4 a2 = *(const float4*)(pA + 8);
    const float4 a3 = *(const float4*)(pA + 12);
    v16bf a;
    a[0]  = (__bf16)a0.x; a[1]  = (__bf16)a0.y; a[2]  = (__bf16)a0.z; a[3]  = (__bf16)a0.w;
    a[4]  = (__bf16)a1.x; a[5]  = (__bf16)a1.y; a[6]  = (__bf16)a1.z; a[7]  = (__bf16)a1.w;
    a[8]  = (__bf16)a2.x; a[9]  = (__bf16)a2.y; a[10] = (__bf16)a2.z; a[11] = (__bf16)a2.w;
    a[12] = (__bf16)a3.x; a[13] = (__bf16)a3.y; a[14] = (__bf16)a3.z; a[15] = (__bf16)a3.w;

    v16bf b0, b1;
#pragma unroll
    for (int e = 0; e < 16; ++e) {      // constant-immediate offsets, coalesced per half
      b0[e] = (__bf16)pB0[e * N];
      b1[e] = (__bf16)pB1[e * N];
    }
    acc0 = __builtin_amdgcn_wmma_f32_16x16x32_bf16(false, a, false, b0, (short)0, acc0, false, false);
    acc1 = __builtin_amdgcn_wmma_f32_16x16x32_bf16(false, a, false, b1, (short)0, acc1, false, false);
    pA  += 32;
    pB0 += (size_t)32 * N;
    pB1 += (size_t)32 * N;
  }

  const float bn0 = (bias && c0 < N) ? bias[c0] : 0.0f;
  const float bn1 = (bias && c1 < N) ? bias[c1] : 0.0f;
#pragma unroll
  for (int i = 0; i < 8; ++i) {
    const size_t row = (size_t)(m0 + i + 8 * h) * N;
    float v0 = acc0[i] + bn0;
    float v1 = acc1[i] + bn1;
    if (relu) { v0 = fmaxf(v0, 0.0f); v1 = fmaxf(v1, 0.0f); }
    if (c0 < N) C[row + c0] = v0;
    if (c1 < N) C[row + c1] = v1;
  }
}

// ---------------- noise injection ----------------
__global__ void qkv_noise_kernel(float* __restrict__ q, float* __restrict__ k,
                                 float* __restrict__ v, float* __restrict__ nq_out,
                                 const float* __restrict__ sigmas)
{
  uint32_t idx = blockIdx.x * 256u + threadIdx.x;   // BP*D = 65536 total
  float n0 = sigmas[0] * gaussian(idx, 0x11u);
  q[idx]      += n0;
  nq_out[idx]  = n0;
  k[idx]      += sigmas[1] * gaussian(idx, 0x22u);
  v[idx]      += sigmas[2] * gaussian(idx, 0x33u);
}

__global__ void z_noise_kernel(float* __restrict__ z, float* __restrict__ nz_out,
                               const float* __restrict__ sigmas)
{
  uint32_t idx = blockIdx.x * 256u + threadIdx.x;
  float n3 = sigmas[3] * gaussian(idx, 0x44u);
  z[idx]      += n3;
  nz_out[idx]  = n3;
}

// ---------------- single-query attention over the t+1 KV window ----------------
__global__ void attention_kernel(const float* __restrict__ q,  const float* __restrict__ kx,
                                 const float* __restrict__ vx, const float* __restrict__ Kin,
                                 const float* __restrict__ Vin, float* __restrict__ attn_out,
                                 float* __restrict__ h_out)
{
  __shared__ float q_s[D_];
  __shared__ float l_s[T_ + 1];
  __shared__ float red[D_];
  const int bp  = blockIdx.x;
  const int tid = threadIdx.x;

  q_s[tid] = q[(size_t)bp * D_ + tid];
  // warm L2/WGP$ for the V rows that the second pass will stream (global_prefetch_b8)
  __builtin_prefetch(Vin + ((size_t)bp * S_ + tid) * D_, 0, 3);
  __syncthreads();

  // logits: s = tid covers rows 0..T_-1 (from Kin); thread 0 also does s = T_ (fresh k)
  {
    const float4* Kr = (const float4*)(Kin + ((size_t)bp * S_ + tid) * D_);
    float acc = 0.f;
#pragma unroll 4
    for (int i = 0; i < D_ / 4; ++i) {
      float4 kv = Kr[i];
      acc += q_s[4*i+0]*kv.x + q_s[4*i+1]*kv.y + q_s[4*i+2]*kv.z + q_s[4*i+3]*kv.w;
    }
    l_s[tid] = acc * SCALE;
  }
  if (tid == 0) {
    const float4* Kr = (const float4*)(kx + (size_t)bp * D_);
    float acc = 0.f;
    for (int i = 0; i < D_ / 4; ++i) {
      float4 kv = Kr[i];
      acc += q_s[4*i+0]*kv.x + q_s[4*i+1]*kv.y + q_s[4*i+2]*kv.z + q_s[4*i+3]*kv.w;
    }
    l_s[T_] = acc * SCALE;
  }
  __syncthreads();

  // softmax over 257 values
  float mv = l_s[tid];
  if (tid == 0) mv = fmaxf(mv, l_s[T_]);
  red[tid] = mv; __syncthreads();
  for (int off = 128; off > 0; off >>= 1) { if (tid < off) red[tid] = fmaxf(red[tid], red[tid + off]); __syncthreads(); }
  const float mx = red[0]; __syncthreads();

  const float ev    = __expf(l_s[tid] - mx);
  const float extra = (tid == 0) ? __expf(l_s[T_] - mx) : 0.f;
  red[tid] = ev + extra; __syncthreads();
  for (int off = 128; off > 0; off >>= 1) { if (tid < off) red[tid] += red[tid + off]; __syncthreads(); }
  const float inv = 1.0f / red[0]; __syncthreads();

  l_s[tid] = ev * inv;
  attn_out[(size_t)bp * (T_ + 1) + tid] = l_s[tid];
  if (tid == 0) {
    l_s[T_] = extra * inv;
    attn_out[(size_t)bp * (T_ + 1) + T_] = l_s[T_];
  }
  __syncthreads();

  // h = attn @ V_window : d-parallel, coalesced across lanes
  float acc = l_s[T_] * vx[(size_t)bp * D_ + tid];
  for (int s = 0; s < T_; ++s)
    acc += l_s[s] * Vin[((size_t)bp * S_ + s) * D_ + tid];
  h_out[(size_t)bp * D_ + tid] = acc;
}

// ---------------- layernorm(a + b) * g + be ----------------
__global__ void ln_residual_kernel(const float* __restrict__ a, const float* __restrict__ b,
                                   const float* __restrict__ g, const float* __restrict__ be,
                                   float* __restrict__ out)
{
  __shared__ float red[D_];
  const int bp = blockIdx.x, tid = threadIdx.x;
  const float x = a[(size_t)bp * D_ + tid] + b[(size_t)bp * D_ + tid];
  red[tid] = x; __syncthreads();
  for (int off = 128; off > 0; off >>= 1) { if (tid < off) red[tid] += red[tid + off]; __syncthreads(); }
  const float mu = red[0] * (1.0f / D_); __syncthreads();
  const float dx = x - mu;
  red[tid] = dx * dx; __syncthreads();
  for (int off = 128; off > 0; off >>= 1) { if (tid < off) red[tid] += red[tid + off]; __syncthreads(); }
  const float var = red[0] * (1.0f / D_); __syncthreads();
  out[(size_t)bp * D_ + tid] = dx * rsqrtf(var + EPS_) * g[tid] + be[tid];
}

// ---------------- per-row vocab softmax -> particle weight at label y ----------------
__global__ void pred_weight_kernel(const float* __restrict__ preds, const int* __restrict__ y,
                                   float* __restrict__ w)
{
  __shared__ float red[256];
  const int bp = blockIdx.x, tid = threadIdx.x;
  const float* row = preds + (size_t)bp * VOCAB_;
  float mx = -3.4e38f;
  for (int i = tid; i < VOCAB_; i += 256) mx = fmaxf(mx, row[i]);
  red[tid] = mx; __syncthreads();
  for (int off = 128; off > 0; off >>= 1) { if (tid < off) red[tid] = fmaxf(red[tid], red[tid + off]); __syncthreads(); }
  mx = red[0]; __syncthreads();
  float s = 0.f;
  for (int i = tid; i < VOCAB_; i += 256) s += __expf(row[i] - mx);
  red[tid] = s; __syncthreads();
  for (int off = 128; off > 0; off >>= 1) { if (tid < off) red[tid] += red[tid + off]; __syncthreads(); }
  if (tid == 0) w[bp] = __expf(row[y[bp]] - mx) / red[0];
}

// ---------------- per-batch particle softmax + categorical sample ----------------
__global__ void sample_kernel(const float* __restrict__ w, int* __restrict__ i_t)
{
  const int tid = threadIdx.x;        // 256 = (b,p)
  const int b = tid >> 4;
  float wv[16];
  float mx = -3.4e38f;
#pragma unroll
  for (int j = 0; j < 16; ++j) { wv[j] = w[b * 16 + j]; mx = fmaxf(mx, wv[j]); }
  float sum = 0.f;
#pragma unroll
  for (int j = 0; j < 16; ++j) { wv[j] = __expf(wv[j] - mx); sum += wv[j]; }
  const float u = u01(hmix(0x5A17C0DEu + (uint32_t)tid * 0x9E3779B9u)) * sum;
  float c = 0.f; int sel = 15;
#pragma unroll
  for (int j = 0; j < 16; ++j) { c += wv[j]; if (u <= c) { sel = j; break; } }
  i_t[tid] = sel;
}

// ---------------- resample gather/copy: dst[b,p,s,:] ----------------
__global__ void resample_kernel(const float* __restrict__ src, const float* __restrict__ trow,
                                float* __restrict__ dst, const int* __restrict__ i_t, int t)
{
  const int s  = blockIdx.x;          // 0..S_-1
  const int bp = blockIdx.y;          // 0..BP-1
  const int b  = bp >> 4;
  const float4* srow;
  if (s < t) {
    const int it = i_t[bp];
    srow = (const float4*)(src + (((size_t)(b * P_ + it)) * S_ + s) * D_);
  } else if (s == t) {
    const int it = i_t[bp];
    srow = (const float4*)(trow + (size_t)(b * P_ + it) * D_);
  } else {
    srow = (const float4*)(src + (((size_t)bp) * S_ + s) * D_);
  }
  float4* drow = (float4*)(dst + (((size_t)bp) * S_ + s) * D_);
  drow[threadIdx.x] = srow[threadIdx.x];
}

// ---------------- host launch ----------------
extern "C" void kernel_launch(void* const* d_in, const int* in_sizes, int n_in,
                              void* d_out, int out_size, void* d_ws, size_t ws_size,
                              hipStream_t stream) {
  (void)in_sizes; (void)n_in; (void)out_size; (void)ws_size;

  const float* x     = (const float*)d_in[0];
  const int*   y     = (const int*)  d_in[1];
  const float* Kin   = (const float*)d_in[2];
  const float* Vin   = (const float*)d_in[3];
  const float* Rin   = (const float*)d_in[4];
  const float* wq    = (const float*)d_in[5];
  const float* bq    = (const float*)d_in[6];
  const float* wk    = (const float*)d_in[7];
  const float* bk    = (const float*)d_in[8];
  const float* wv    = (const float*)d_in[9];
  const float* bv    = (const float*)d_in[10];
  const float* wo    = (const float*)d_in[11];
  const float* bo    = (const float*)d_in[12];
  const float* w1    = (const float*)d_in[13];
  const float* b1    = (const float*)d_in[14];
  const float* w2    = (const float*)d_in[15];
  const float* b2    = (const float*)d_in[16];
  const float* ln1_g = (const float*)d_in[17];
  const float* ln1_b = (const float*)d_in[18];
  const float* ln2_g = (const float*)d_in[19];
  const float* ln2_b = (const float*)d_in[20];
  const float* w_out = (const float*)d_in[21];
  const float* sig   = (const float*)d_in[22];
  // d_in[23] = timestep (== T_ = 256 by construction; grid shapes need it on host)

  float* out = (float*)d_out;
  float* ws  = (float*)d_ws;

  float* q_s    = ws + WS_Q;
  float* k_s    = ws + WS_KS;
  float* v_s    = ws + WS_VS;
  float* h_s    = ws + WS_H;
  float* z_s    = ws + WS_Z;
  float* o1_s   = ws + WS_O1;
  float* f1_s   = ws + WS_F1;
  float* f2_s   = ws + WS_F2;
  float* pred_s = ws + WS_PRED;
  float* w_s    = ws + WS_W;
  int*   it_s   = (int*)(ws + WS_IT);

  const dim3 gD((D_    + 127) / 128, BP / 16);   // N=256
  const dim3 gF((DFF_  + 127) / 128, BP / 16);   // N=1024
  const dim3 gV((VOCAB_+ 127) / 128, BP / 16);   // N=5000 (guarded)

  // 1) QKV projections (WMMA bf16)
  gemm_wmma_bf16<D_, D_><<<gD, 128, 0, stream>>>(x, wq, bq, q_s, 0);
  gemm_wmma_bf16<D_, D_><<<gD, 128, 0, stream>>>(x, wk, bk, k_s, 0);
  gemm_wmma_bf16<D_, D_><<<gD, 128, 0, stream>>>(x, wv, bv, v_s, 0);

  // 2) inject q/k/v noise, emit noise_q
  qkv_noise_kernel<<<BP, 256, 0, stream>>>(q_s, k_s, v_s, out + OFF_NQ, sig);

  // 3) attention (emits attn, h = attn @ V_window)
  attention_kernel<<<BP, 256, 0, stream>>>(q_s, k_s, v_s, Kin, Vin, out + OFF_ATTN, h_s);

  // 4) output projection + z noise + LN1
  gemm_wmma_bf16<D_, D_><<<gD, 128, 0, stream>>>(h_s, wo, bo, z_s, 0);
  z_noise_kernel<<<BP, 256, 0, stream>>>(z_s, out + OFF_NZ, sig);
  ln_residual_kernel<<<BP, 256, 0, stream>>>(z_s, x, ln1_g, ln1_b, o1_s);

  // 5) FFN + LN2 -> r (written straight into d_out)
  gemm_wmma_bf16<D_, DFF_><<<gF, 128, 0, stream>>>(o1_s, w1, b1, f1_s, 1);
  gemm_wmma_bf16<DFF_, D_><<<gD, 128, 0, stream>>>(f1_s, w2, b2, f2_s, 0);
  ln_residual_kernel<<<BP, 256, 0, stream>>>(f2_s, o1_s, ln2_g, ln2_b, out + OFF_R);

  // 6) vocab head -> per-particle weights -> resampling indices
  gemm_wmma_bf16<D_, VOCAB_><<<gV, 128, 0, stream>>>(out + OFF_R, w_out, nullptr, pred_s, 0);
  pred_weight_kernel<<<BP, 256, 0, stream>>>(pred_s, y, w_s);
  sample_kernel<<<1, 256, 0, stream>>>(w_s, it_s);

  // 7) resample K, V, R (gather for s<=t, passthrough for s>t)
  const dim3 gR(S_, BP);
  resample_kernel<<<gR, 64, 0, stream>>>(Kin, k_s,          out + OFF_K,  it_s, T_);
  resample_kernel<<<gR, 64, 0, stream>>>(Vin, v_s,          out + OFF_V,  it_s, T_);
  resample_kernel<<<gR, 64, 0, stream>>>(Rin, out + OFF_R,  out + OFF_RR, it_s, T_);
}
